// Encoder_73710228734625
// MI455X (gfx1250) — matrix operand order repaired and checked by reading, not compiled, not run on hardware
//
#include <hip/hip_runtime.h>

typedef _Float16 h16;
typedef __attribute__((ext_vector_type(8)))  _Float16 v8h;
typedef __attribute__((ext_vector_type(16))) _Float16 v16h;
typedef __attribute__((ext_vector_type(8)))  float    v8f;

#define EMBD   256
#define UNITS  512
#define BATCH  32
#define SEQ    512
#define NGATE  2048            // 4*UNITS
#define NTT    128             // NGATE/16  (N tiles)

// ---------------------------------------------------------------------------
// WMMA helpers (CDNA5 wave32, V_WMMA_F32_16X16X32_F16)
// ---------------------------------------------------------------------------
__device__ __forceinline__ v8f wmma_f16(v16h a, v16h b, v8f c) {
  return __builtin_amdgcn_wmma_f32_16x16x32_f16(
      /*neg_a=*/false, a, /*neg_b=*/false, b,
      /*c_mod=*/(short)0, c, /*reuse_a=*/false, /*reuse_b=*/false);
}

// A fragment (16x32 f16) from a row-major [rows x ld] f16 matrix.
// ISA layout: lanes 0-15 -> row M=lane, K = k0+{0..7,16..23};
//             lanes 16-31 -> row M=lane-16, K = k0+{8..15,24..31}
__device__ __forceinline__ v16h load_a_frag(const h16* __restrict__ A,
                                            int row0, int k0, int ld) {
  const int lane = threadIdx.x & 31;
  const int row  = row0 + (lane & 15);
  const int kb   = k0 + ((lane >> 4) << 3);
  const h16* p = A + (size_t)row * ld + kb;
  v8h lo = *(const v8h*)p;          // K = kb .. kb+7     (16B aligned)
  v8h hi = *(const v8h*)(p + 16);   // K = kb+16 .. kb+23
  v16h a;
#pragma unroll
  for (int i = 0; i < 8; ++i) { a[i] = lo[i]; a[8 + i] = hi[i]; }
  return a;
}

// B fragment from pre-packed layout: contiguous 16 halves per lane.
__device__ __forceinline__ v16h load_b_pack(const h16* __restrict__ P,
                                            int kt, int nt) {
  const int lane = threadIdx.x & 31;
  return *(const v16h*)(P + ((((size_t)kt * NTT + nt) * 32 + lane) << 4));
}

// ---------------------------------------------------------------------------
// Pack f32 weight [K x NGATE] into B-fragment order, cast to f16.
// dst[((kt*NTT+nt)*32+lane)*16 + h] = W[kt*32 + ((lane>>4)<<3) + (h&7) + ((h>>3)<<4)][nt*16 + (lane&15)]
// ---------------------------------------------------------------------------
__global__ void pack_b_kernel(const float* __restrict__ src,
                              h16* __restrict__ dst, int ktiles) {
  size_t i = (size_t)blockIdx.x * blockDim.x + threadIdx.x;
  size_t total = (size_t)ktiles * NTT * 32 * 16;
  if (i >= total) return;
  int hh   = (int)(i & 15);
  int lane = (int)((i >> 4) & 31);
  int tidx = (int)(i >> 9);
  int nt = tidx % NTT;
  int kt = tidx / NTT;
  int k = kt * 32 + ((lane >> 4) << 3) + (hh & 7) + ((hh >> 3) << 4);
  int n = nt * 16 + (lane & 15);
  dst[i] = (h16)src[(size_t)k * NGATE + n];
}

// ---------------------------------------------------------------------------
// Embedding gather + f32 -> f16 cast.  grid = B*S blocks, 256 threads.
// ---------------------------------------------------------------------------
__global__ void embed_kernel(const int* __restrict__ x,
                             const float* __restrict__ emb,
                             h16* __restrict__ xe) {
  const int bs = blockIdx.x;       // 0..B*S-1
  const int e  = threadIdx.x;      // 0..EMBD-1
  const int tok = x[bs];
  xe[(size_t)bs * EMBD + e] = (h16)emb[(size_t)tok * EMBD + e];
}

// ---------------------------------------------------------------------------
// Input projection GEMM: zx[r,n] = xe[r,:] @ W[:,n] + bias[n], stored f16.
// M = B*S = 16384, K = EMBD = 256, N = NGATE = 2048.
// 256 threads = 8 waves/block, one 16x16 tile per wave.
// grid = (M/16 * N/16 / 8, 2 directions)
// ---------------------------------------------------------------------------
__global__ void ingemm_kernel(const h16* __restrict__ xe,
                              const h16* __restrict__ wpF,
                              const h16* __restrict__ wpB,
                              const float* __restrict__ biasF,
                              const float* __restrict__ biasB,
                              h16* __restrict__ zxF,
                              h16* __restrict__ zxB) {
  const int dir = blockIdx.y;
  const h16*   wp   = dir ? wpB : wpF;
  const float* bias = dir ? biasB : biasF;
  h16*         zx   = dir ? zxB : zxF;

  const int wave = threadIdx.x >> 5;
  const int lane = threadIdx.x & 31;
  const int tile = blockIdx.x * 8 + wave;
  const int mt = tile / NTT;
  const int nt = tile % NTT;

  v8f acc = {};
#pragma unroll
  for (int kt = 0; kt < EMBD / 32; ++kt) {
    v16h a = load_a_frag(xe, mt * 16, kt * 32, EMBD);
    v16h b = load_b_pack(wp, kt, nt);
    acc = wmma_f16(a, b, acc);
  }
  const int col   = nt * 16 + (lane & 15);
  const float bv  = bias[col];
  const int rbase = mt * 16 + ((lane >> 4) << 3);
#pragma unroll
  for (int v = 0; v < 8; ++v)
    zx[(size_t)(rbase + v) * NGATE + col] = (h16)(acc[v] + bv);
}

// ---------------------------------------------------------------------------
// Persistent bidirectional LSTM scan: 1 workgroup (32 waves) per direction.
// LDS: h (f16, 32KB) + c (f32, 64KB) + z pre-activations (f16, 128KB) = 224KB.
// Per step: [32x512] @ [512x2048] via WMMA (h from LDS, U packed from L2),
// add zx, barrier, gate math, barrier.
// ---------------------------------------------------------------------------
__global__ void __launch_bounds__(1024)
lstm_scan_kernel(const h16* __restrict__ zxF, const h16* __restrict__ zxB,
                 const h16* __restrict__ upF, const h16* __restrict__ upB,
                 const float* __restrict__ fh, const float* __restrict__ fc,
                 const float* __restrict__ bh, const float* __restrict__ bc,
                 float* __restrict__ out) {
  const int dir = blockIdx.x;                 // 0 = forward, 1 = backward
  const h16*   zx = dir ? zxB : zxF;
  const h16*   up = dir ? upB : upF;
  const float* h0 = dir ? bh : fh;
  const float* c0 = dir ? bc : fc;
  float* hT = out + (size_t)BATCH * SEQ * (2 * UNITS) + (size_t)dir * 2 * BATCH * UNITS;
  float* cT = hT + BATCH * UNITS;

  extern __shared__ unsigned char smem[];
  h16*   h_sh = (h16*)smem;                                     // [32][512]
  float* c_sh = (float*)(smem + BATCH * UNITS * 2);             // [32][512]
  h16*   z_sh = (h16*)(smem + BATCH * UNITS * 2 + BATCH * UNITS * 4); // [32][2048]

  const int tid = threadIdx.x;
  for (int e = tid; e < BATCH * UNITS; e += 1024) {
    h_sh[e] = (h16)h0[e];
    c_sh[e] = c0[e];
  }
  __syncthreads();

  const int wave   = tid >> 5;
  const int lane   = tid & 31;
  const int ntBase = wave * 4;                // 4 N-tiles x 2 M-tiles per wave

  for (int t = 0; t < SEQ; ++t) {
    const int s = dir ? (SEQ - 1 - t) : t;

    v8f acc[2][4] = {};
    for (int kt = 0; kt < UNITS / 32; ++kt) {
      v16h a0 = load_a_frag(h_sh, 0,  kt * 32, UNITS);
      v16h a1 = load_a_frag(h_sh, 16, kt * 32, UNITS);
#pragma unroll
      for (int j = 0; j < 4; ++j) {
        v16h b = load_b_pack(up, kt, ntBase + j);
        acc[0][j] = wmma_f16(a0, b, acc[0][j]);
        acc[1][j] = wmma_f16(a1, b, acc[1][j]);
      }
    }

    // z = h@U + zx  -> LDS (f16 pre-activations)
#pragma unroll
    for (int m = 0; m < 2; ++m) {
      const int rb = m * 16 + ((lane >> 4) << 3);
#pragma unroll
      for (int j = 0; j < 4; ++j) {
        const int col = (ntBase + j) * 16 + (lane & 15);
#pragma unroll
        for (int v = 0; v < 8; ++v) {
          const int b = rb + v;               // batch row
          float zv = acc[m][j][v] + (float)zx[((size_t)b * SEQ + s) * NGATE + col];
          z_sh[b * NGATE + col] = (h16)zv;
        }
      }
    }
    __syncthreads();

    // Gates: i,f,g,o split of NGATE; update c,h; emit h to output.
    for (int e = tid; e < BATCH * UNITS; e += 1024) {
      const int b = e >> 9;
      const int u = e & 511;
      const h16* zrow = z_sh + b * NGATE;
      const float zi = (float)zrow[u];
      const float zf = (float)zrow[u + 512];
      const float zg = (float)zrow[u + 1024];
      const float zo = (float)zrow[u + 1536];
      const float ig = 1.0f / (1.0f + __expf(-zi));
      const float fg = 1.0f / (1.0f + __expf(-zf));
      const float gg = tanhf(zg);
      const float og = 1.0f / (1.0f + __expf(-zo));
      const float cv = fg * c_sh[e] + ig * gg;
      const float hv = og * tanhf(cv);
      c_sh[e] = cv;
      h_sh[e] = (h16)hv;
      out[((size_t)b * SEQ + s) * (2 * UNITS) + dir * UNITS + u] = hv;
    }
    __syncthreads();
  }

  for (int e = tid; e < BATCH * UNITS; e += 1024) {
    hT[e] = (float)h_sh[e];
    cT[e] = c_sh[e];
  }
}

// ---------------------------------------------------------------------------
// Host launcher
// ---------------------------------------------------------------------------
extern "C" void kernel_launch(void* const* d_in, const int* in_sizes, int n_in,
                              void* d_out, int out_size, void* d_ws, size_t ws_size,
                              hipStream_t stream) {
  const int*   x    = (const int*)d_in[0];
  const float* fh   = (const float*)d_in[1];
  const float* fc   = (const float*)d_in[2];
  const float* bh   = (const float*)d_in[3];
  const float* bc   = (const float*)d_in[4];
  const float* emb  = (const float*)d_in[5];
  const float* Wf   = (const float*)d_in[6];
  const float* Uf   = (const float*)d_in[7];
  const float* bfv  = (const float*)d_in[8];
  const float* Wb   = (const float*)d_in[9];
  const float* Ub   = (const float*)d_in[10];
  const float* bbv  = (const float*)d_in[11];
  float* outp = (float*)d_out;

  // Workspace layout (all 32B-aligned):
  char* ws = (char*)d_ws;
  h16* xe16 = (h16*)(ws);                        //  8,388,608 B  [16384 x 256] f16
  h16* wpF  = (h16*)(ws + 8388608);              //  1,048,576 B  packed Wf
  h16* wpB  = (h16*)(ws + 9437184);              //  1,048,576 B  packed Wb
  h16* upF  = (h16*)(ws + 10485760);             //  2,097,152 B  packed Uf
  h16* upB  = (h16*)(ws + 12582912);             //  2,097,152 B  packed Ub
  h16* zxF  = (h16*)(ws + 14680064);             // 67,108,864 B  [16384 x 2048] f16
  h16* zxB  = (h16*)(ws + 81788928);             // 67,108,864 B
  // total: 148,897,792 bytes

  // 1) Pack weights into WMMA B-fragment order (f16).
  pack_b_kernel<<<2048, 256, 0, stream>>>(Wf, wpF, EMBD / 32);
  pack_b_kernel<<<2048, 256, 0, stream>>>(Wb, wpB, EMBD / 32);
  pack_b_kernel<<<4096, 256, 0, stream>>>(Uf, upF, UNITS / 32);
  pack_b_kernel<<<4096, 256, 0, stream>>>(Ub, upB, UNITS / 32);

  // 2) Embedding gather + cast.
  embed_kernel<<<BATCH * SEQ, EMBD, 0, stream>>>(x, emb, xe16);

  // 3) Input projection GEMMs (both directions), bias fused, f16 output.
  ingemm_kernel<<<dim3((BATCH * SEQ / 16) * NTT / 8, 2), 256, 0, stream>>>(
      xe16, wpF, wpB, bfv, bbv, zxF, zxB);

  // 4) Persistent recurrent scan: 1 workgroup per direction, 224KB LDS.
  const size_t lds_bytes = (size_t)BATCH * UNITS * 2   // h f16
                         + (size_t)BATCH * UNITS * 4   // c f32
                         + (size_t)BATCH * NGATE * 2;  // z f16
  lstm_scan_kernel<<<2, 1024, lds_bytes, stream>>>(zxF, zxB, upF, upB,
                                                   fh, fc, bh, bc, outp);
}
